// GraphAttention_Naive_44616120271219
// MI455X (gfx1250) — matrix operand order repaired and checked by reading, not compiled, not run on hardware
//
#include <hip/hip_runtime.h>

typedef __attribute__((ext_vector_type(16))) __bf16 v16bf;
typedef __attribute__((ext_vector_type(8)))  float  v8f;
typedef __attribute__((ext_vector_type(4)))  int    v4i;
typedef __attribute__((ext_vector_type(8)))  int    v8i;

#define BATCH  2
#define NSEQ   2048
#define DMODEL 512
#define NHEAD  8
#define DHEAD  512
#define FDIM   4096
#define MROWS  (BATCH * NSEQ)   // 4096

// padded LDS tile strides (elements) -> conflict-free ds_load_b128 fragments
#define KT_LD  520   // 512 + 8  (8 elem = 4 dwords pad per row)
#define VT_LD  40    // 32 + 8

union FragU { uint4 q[2]; v16bf v; };

// 16-bit A-matrix 16x32 fragment: lane holds row (lane&15); K runs
// [khalf*16, +16) contiguous where khalf = lane>>4. Two 16B loads.
static __device__ __forceinline__ v16bf load_a_frag(const __bf16* rowk, int khalf) {
  FragU u;
  u.q[0] = *(const uint4*)(rowk + khalf * 16);
  u.q[1] = *(const uint4*)(rowk + khalf * 16 + 8);
  return u.v;
}
// 16-bit B-matrix 32x16 fragment: lane holds column (lane&15); K values
// (lane>>4)*16 + e, e=0..15 contiguous. Caller pre-adds (lane>>4)*16.
static __device__ __forceinline__ v16bf load_b_frag(const __bf16* colk) {
  FragU u;
  u.q[0] = *(const uint4*)(colk);
  u.q[1] = *(const uint4*)(colk + 8);
  return u.v;
}
static __device__ __forceinline__ v8f wmma_bf16(v16bf a, v16bf b, v8f c) {
  return __builtin_amdgcn_wmma_f32_16x16x32_bf16(false, a, false, b, (short)0, c,
                                                 false, false);
}

// ---- Tensor Data Mover: DMA one 2D bf16 tile (global -> LDS), TENSORcnt ----
// D# per CDNA5 ISA ch.8: group0 {count, lds_addr, global_addr, type=2},
// group1 {data_size=1(2B), pad cfg, tensor dims, tile dims, dim0 stride},
// groups 2/3 zero (dims >2 unused).  pad_interval enc: 3->16 dw, 7->256 dw;
// pad_amount enc: 3 -> 4 dwords (8 bf16) appended per interval.
static __device__ __forceinline__ void tdm_load_2d_bf16(
    unsigned lds_addr, unsigned long long gaddr,
    unsigned tile_d0, unsigned tile_d1,
    unsigned long long stride0_elems,
    unsigned pad_interval, unsigned pad_amount) {
  v4i g0; v8i g1; v4i g2; v4i g3;
  g0[0] = 1;                                   // count=1 (valid user D#)
  g0[1] = (int)lds_addr;                       // LDS byte address
  g0[2] = (int)(unsigned)gaddr;                // global_addr[31:0]
  g0[3] = (int)((((unsigned)(gaddr >> 32)) & 0x01FFFFFFu) | (2u << 30)); // +type=2
  g1[0] = (int)((1u << 16) | (1u << 20) | (pad_interval << 22) | (pad_amount << 25));
  g1[1] = (int)((tile_d0 & 0xFFFFu) << 16);    // tensor_dim0[15:0] (== tile_d0)
  g1[2] = (int)(((tile_d0 >> 16) & 0xFFFFu) | ((tile_d1 & 0xFFFFu) << 16)); // dim0 hi | dim1 lo
  g1[3] = (int)(((tile_d1 >> 16) & 0xFFFFu) | ((tile_d0 & 0xFFFFu) << 16)); // dim1 hi | tile_dim0
  g1[4] = (int)(tile_d1 & 0xFFFFu);            // tile_dim1 ; tile_dim2 = 0
  g1[5] = (int)(unsigned)stride0_elems;        // tensor_dim0_stride[31:0]
  g1[6] = (int)((unsigned)(stride0_elems >> 32) & 0xFFFFu); // stride[47:32]
  g1[7] = 0;
  g2 = (v4i)0;
  g3 = (v4i)0;
  asm volatile("tensor_load_to_lds %0, %1, %2, %3"
               :: "s"(g0), "s"(g1), "s"(g2), "s"(g3) : "memory");
}

// generic LDS byte offset (flat LDS aperture: addr[31:0] is the LDS offset)
static __device__ __forceinline__ unsigned lds_off(const void* p) {
  return (unsigned)(unsigned long long)(uintptr_t)p;
}

// ---------------- elementwise conversion kernels ----------------
__global__ void k_cvt(const float* __restrict__ src, __bf16* __restrict__ dst, int n) {
  int i = blockIdx.x * blockDim.x + threadIdx.x;
  if (i < n) dst[i] = (__bf16)src[i];
}

// W is [K, N] row-major fp32 -> Wt [N, K] bf16 (K contiguous for B-fragments)
__global__ void k_wt(const float* __restrict__ W, __bf16* __restrict__ Wt, int K, int N) {
  int i = blockIdx.x * blockDim.x + threadIdx.x;
  if (i < N * K) {
    int n = i / K, k = i - n * K;
    Wt[i] = (__bf16)W[(size_t)k * N + n];
  }
}

// ---------------- fused GEMM (projections + output) ----------------
// MODE 0: RoPE epilogue, store bf16 row-major [m*FDIM + f]   (Q, K)
// MODE 1: +bias, store bf16 transposed [ (b*FDIM + f)*NSEQ + pos ]  (V)
// MODE 2: +bias, store fp32 row-major [m*Ncols + f]          (final @ Wo)
template <int MODE>
__global__ __launch_bounds__(256)
void k_gemm(const __bf16* __restrict__ A, const __bf16* __restrict__ Bt,
            void* __restrict__ outp,
            const float* __restrict__ ctab, const float* __restrict__ stab,
            const float* __restrict__ bias, int K, int Ncols) {
  const int lane  = threadIdx.x & 31;
  const int wave  = threadIdx.x >> 5;
  const int l16   = lane & 15;
  const int khalf = lane >> 4;
  const int row0  = blockIdx.x * 64 + (wave >> 2) * 32;
  const int col0  = blockIdx.y * 128 + (wave & 3) * 32;

  v8f C[2][2] = {};
  const __bf16* a0 = A + (size_t)(row0 + l16) * K;
  const __bf16* a1 = a0 + (size_t)16 * K;
  const __bf16* b0 = Bt + (size_t)(col0 + l16) * K + khalf * 16;
  const __bf16* b1 = b0 + (size_t)16 * K;

  for (int k0 = 0; k0 < K; k0 += 32) {
    v16bf A0 = load_a_frag(a0 + k0, khalf);
    v16bf A1 = load_a_frag(a1 + k0, khalf);
    v16bf B0 = load_b_frag(b0 + k0);
    v16bf B1 = load_b_frag(b1 + k0);
    C[0][0] = wmma_bf16(A0, B0, C[0][0]);
    C[0][1] = wmma_bf16(A0, B1, C[0][1]);
    C[1][0] = wmma_bf16(A1, B0, C[1][0]);
    C[1][1] = wmma_bf16(A1, B1, C[1][1]);
  }

#pragma unroll
  for (int i = 0; i < 2; ++i)
#pragma unroll
    for (int j = 0; j < 2; ++j)
#pragma unroll
      for (int v = 0; v < 8; ++v) {
        int   m   = row0 + i * 16 + v + 8 * khalf;
        int   f   = col0 + j * 16 + l16;
        float val = C[i][j][v];
        if constexpr (MODE == 0) {
          // f parity == lane parity: partner value lives on lane^1
          float other = __shfl_xor(val, 1, 32);
          int   pos   = m & (NSEQ - 1);
          float cc    = ctab[(size_t)pos * FDIM + f];
          float ss    = stab[(size_t)pos * FDIM + f];
          float rot   = (f & 1) ? other : -other;  // (x0,x1)->(-x1,x0)
          ((__bf16*)outp)[(size_t)m * FDIM + f] = (__bf16)(val * cc + rot * ss);
        } else if constexpr (MODE == 1) {
          int bb  = m >> 11;            // m / NSEQ
          int pos = m & (NSEQ - 1);
          float r = val + bias[f];
          ((__bf16*)outp)[((size_t)bb * FDIM + f) * NSEQ + pos] = (__bf16)r;
        } else {
          ((float*)outp)[(size_t)m * Ncols + f] = val + bias[f];
        }
      }
}

// ---------------- flash attention (causal, per-head) ----------------
// block: 256 threads = 8 waves. One 16-row query tile per block;
// each wave owns a 64-wide slice of d_head (8*64 = 512).
// K/V tiles are DMA'd into LDS by the Tensor Data Mover (TENSORcnt).
__global__ __launch_bounds__(256)
void k_attn(const __bf16* __restrict__ Qb, const __bf16* __restrict__ Kb,
            const __bf16* __restrict__ Vt, float* __restrict__ yh) {
  __shared__ __bf16 Ktile[32][KT_LD];   // 32 keys x 512 dh (padded)  ~33.3 KB
  __shared__ __bf16 Vtile[512][VT_LD];  // 512 dh  x 32 keys (padded) ~41 KB
  __shared__ float  Sbuf[8][16][32];    // per-wave partial scores     16 KB
  __shared__ __bf16 Pbuf[16][32];       // probs, reshaped to A-frag    1 KB

  const int lane  = threadIdx.x & 31;
  const int w     = threadIdx.x >> 5;
  const int l16   = lane & 15;
  const int khalf = lane >> 4;
  const int q0    = blockIdx.x * 16;
  const int bh    = blockIdx.y;        // b*NHEAD + h
  const int b     = bh >> 3;
  const int h     = bh & 7;
  const size_t headOff = (size_t)b * NSEQ * FDIM + (size_t)h * DHEAD;
  const __bf16* Vhead  = Vt + ((size_t)b * FDIM + (size_t)h * DHEAD) * NSEQ;
  const int d0    = w * 64;            // this wave's d_head slice

  v16bf Qf[2];
#pragma unroll
  for (int kk = 0; kk < 2; ++kk)
    Qf[kk] = load_a_frag(Qb + headOff + (size_t)(q0 + l16) * FDIM + d0 + kk * 32, khalf);

  v8f   O[4] = {};
  float mrow[8], lrow[8];
#pragma unroll
  for (int v = 0; v < 8; ++v) { mrow[v] = -1e30f; lrow[v] = 0.f; }

  for (int j0 = 0; j0 < q0 + 16; j0 += 32) {   // causal: keys <= q0+15
    __syncthreads();   // previous tile's LDS fragment reads are complete
    if (w == 0) {
      // K tile: 32 rows of 512 bf16, row stride FDIM; pad each row (256 dw)
      // by 4 dwords -> LDS stride 520 elements.
      tdm_load_2d_bf16(lds_off(&Ktile[0][0]),
                       (unsigned long long)(uintptr_t)(Kb + headOff + (size_t)j0 * FDIM),
                       512u, 32u, (unsigned long long)FDIM, 7u, 3u);
      // V tile: 512 rows of 32 bf16, row stride NSEQ; pad each row (16 dw)
      // by 4 dwords -> LDS stride 40 elements.
      tdm_load_2d_bf16(lds_off(&Vtile[0][0]),
                       (unsigned long long)(uintptr_t)(Vhead + j0),
                       32u, 512u, (unsigned long long)NSEQ, 3u, 3u);
      __builtin_amdgcn_s_wait_tensorcnt(0);
    }
    __syncthreads();   // K/V tiles resident in LDS

    // ---- partial S = Q * K^T over this wave's 64 dh (B-frags from LDS) ----
    v8f Sp[2] = {};
#pragma unroll
    for (int kk = 0; kk < 2; ++kk)
#pragma unroll
      for (int c = 0; c < 2; ++c) {
        v16bf Kf = load_b_frag(&Ktile[c * 16 + l16][d0 + kk * 32 + khalf * 16]);
        Sp[c] = wmma_bf16(Qf[kk], Kf, Sp[c]);
      }
#pragma unroll
    for (int c = 0; c < 2; ++c)
#pragma unroll
      for (int v = 0; v < 8; ++v)
        Sbuf[w][v + 8 * khalf][c * 16 + l16] = Sp[c][v];
    __syncthreads();

    // ---- every wave reduces the 8 partials to the full 16x32 S tile ----
    float S[2][8];
#pragma unroll
    for (int c = 0; c < 2; ++c)
#pragma unroll
      for (int v = 0; v < 8; ++v) {
        float s = 0.f;
#pragma unroll
        for (int ww = 0; ww < 8; ++ww) s += Sbuf[ww][v + 8 * khalf][c * 16 + l16];
        S[c][v] = s;
      }
    if (j0 + 31 > q0) {  // tile crosses the diagonal
#pragma unroll
      for (int c = 0; c < 2; ++c)
#pragma unroll
        for (int v = 0; v < 8; ++v)
          if (j0 + c * 16 + l16 > q0 + v + 8 * khalf) S[c][v] = -1e30f;
    }

    // ---- online softmax (row stats via intra-16-lane xor reductions) ----
    float P[2][8], alpha[8];
#pragma unroll
    for (int v = 0; v < 8; ++v) {
      float t = fmaxf(S[0][v], S[1][v]);
      t = fmaxf(t, __shfl_xor(t, 1, 32));
      t = fmaxf(t, __shfl_xor(t, 2, 32));
      t = fmaxf(t, __shfl_xor(t, 4, 32));
      t = fmaxf(t, __shfl_xor(t, 8, 32));
      float mnew = fmaxf(mrow[v], t);
      alpha[v]   = __expf(mrow[v] - mnew);
      float p0   = __expf(S[0][v] - mnew);
      float p1   = __expf(S[1][v] - mnew);
      float rs   = p0 + p1;
      rs += __shfl_xor(rs, 1, 32);
      rs += __shfl_xor(rs, 2, 32);
      rs += __shfl_xor(rs, 4, 32);
      rs += __shfl_xor(rs, 8, 32);
      lrow[v] = lrow[v] * alpha[v] + rs;
      mrow[v] = mnew;
      P[0][v] = p0;
      P[1][v] = p1;
    }
    __syncthreads();               // Sbuf reads + previous Pbuf use complete
    if (w == 0) {
#pragma unroll
      for (int c = 0; c < 2; ++c)
#pragma unroll
        for (int v = 0; v < 8; ++v)
          Pbuf[v + 8 * khalf][c * 16 + l16] = (__bf16)P[c][v];
    }
#pragma unroll
    for (int c = 0; c < 4; ++c)
#pragma unroll
      for (int v = 0; v < 8; ++v) O[c][v] *= alpha[v];
    __syncthreads();               // Pbuf ready

    // ---- O += P * V over this wave's dh slice (V tile from LDS) ----
    v16bf Pf = load_a_frag(&Pbuf[l16][0], khalf);
#pragma unroll
    for (int c = 0; c < 4; ++c) {
      v16bf Vf = load_b_frag(&Vtile[d0 + c * 16 + l16][khalf * 16]);
      O[c] = wmma_bf16(Pf, Vf, O[c]);
    }
  }

  // ---- normalize and write per-head output [B,H,N,DH] fp32 ----
#pragma unroll
  for (int c = 0; c < 4; ++c)
#pragma unroll
    for (int v = 0; v < 8; ++v) {
      float r = O[c][v] / lrow[v];
      yh[((size_t)bh * NSEQ + q0 + v + 8 * khalf) * DHEAD + d0 + c * 16 + l16] = r;
    }
}

// ---------------- head-sum + QuickGELU, emit bf16 for final GEMM ----------------
__global__ void k_gelu_sum(const float* __restrict__ yh, __bf16* __restrict__ g) {
  int i = blockIdx.x * blockDim.x + threadIdx.x;
  if (i >= BATCH * NSEQ * DMODEL) return;
  int d   = i & (DMODEL - 1);
  int pos = (i >> 9) & (NSEQ - 1);
  int b   = i >> 20;
  float s = 0.f;
#pragma unroll
  for (int h = 0; h < NHEAD; ++h)
    s += yh[((size_t)(b * NHEAD + h) * NSEQ + pos) * DHEAD + d];
  float y = s / (1.f + __expf(-1.702f * s));  // s * sigmoid(1.702 s)
  g[i] = (__bf16)y;
}

// ---------------- launch ----------------
extern "C" void kernel_launch(void* const* d_in, const int* in_sizes, int n_in,
                              void* d_out, int out_size, void* d_ws, size_t ws_size,
                              hipStream_t stream) {
  const float* x    = (const float*)d_in[0];
  const float* ctab = (const float*)d_in[1];
  const float* stab = (const float*)d_in[2];
  // d_in[3] = target_mask (bool causal) -- handled analytically
  const float* Wq = (const float*)d_in[4];
  const float* Wk = (const float*)d_in[5];
  const float* Wv = (const float*)d_in[6];
  const float* bv = (const float*)d_in[7];
  const float* Wo = (const float*)d_in[8];
  const float* bo = (const float*)d_in[9];

  char*  ws  = (char*)d_ws;
  size_t off = 0;
  auto take = [&](size_t bytes) {
    char* p = ws + off;
    off += (bytes + 255) & ~(size_t)255;
    return p;
  };
  __bf16* xb  = (__bf16*)take((size_t)MROWS * DMODEL * 2);
  __bf16* Wqt = (__bf16*)take((size_t)FDIM * DMODEL * 2);
  __bf16* Wkt = (__bf16*)take((size_t)FDIM * DMODEL * 2);
  __bf16* Wvt = (__bf16*)take((size_t)FDIM * DMODEL * 2);
  __bf16* Wot = (__bf16*)take((size_t)DMODEL * DMODEL * 2);
  __bf16* Qb  = (__bf16*)take((size_t)MROWS * FDIM * 2);
  __bf16* Kb  = (__bf16*)take((size_t)MROWS * FDIM * 2);
  __bf16* Vtr = (__bf16*)take((size_t)BATCH * FDIM * NSEQ * 2);
  float*  yh  = (float*)take((size_t)BATCH * NHEAD * NSEQ * DHEAD * 4);
  __bf16* g   = (__bf16*)take((size_t)MROWS * DMODEL * 2);

  k_cvt<<<(MROWS * DMODEL + 255) / 256, 256, 0, stream>>>(x, xb, MROWS * DMODEL);
  k_wt<<<(FDIM * DMODEL + 255) / 256, 256, 0, stream>>>(Wq, Wqt, DMODEL, FDIM);
  k_wt<<<(FDIM * DMODEL + 255) / 256, 256, 0, stream>>>(Wk, Wkt, DMODEL, FDIM);
  k_wt<<<(FDIM * DMODEL + 255) / 256, 256, 0, stream>>>(Wv, Wvt, DMODEL, FDIM);
  k_wt<<<(DMODEL * DMODEL + 255) / 256, 256, 0, stream>>>(Wo, Wot, DMODEL, DMODEL);

  dim3 gproj(MROWS / 64, FDIM / 128);
  k_gemm<0><<<gproj, 256, 0, stream>>>(xb, Wqt, Qb, ctab, stab, nullptr, DMODEL, FDIM);
  k_gemm<0><<<gproj, 256, 0, stream>>>(xb, Wkt, Kb, ctab, stab, nullptr, DMODEL, FDIM);
  k_gemm<1><<<gproj, 256, 0, stream>>>(xb, Wvt, Vtr, nullptr, nullptr, bv, DMODEL, FDIM);

  dim3 gattn(NSEQ / 16, BATCH * NHEAD);
  k_attn<<<gattn, 256, 0, stream>>>(Qb, Kb, Vtr, yh);

  k_gelu_sum<<<(MROWS * DMODEL + 255) / 256, 256, 0, stream>>>(yh, g);

  dim3 gout(MROWS / 64, DMODEL / 128);
  k_gemm<2><<<gout, 256, 0, stream>>>(g, Wot, d_out, nullptr, nullptr, bo, DMODEL, DMODEL);
}
